// NequipLayer_42700564856853
// MI455X (gfx1250) — matrix operand order repaired and checked by reading, not compiled, not run on hardware
//
#include <hip/hip_runtime.h>
#include <hip/hip_bf16.h>

typedef __attribute__((ext_vector_type(16))) _Float16 v16h;
typedef __attribute__((ext_vector_type(8)))  float    v8f;

#define N_NODES 50000
#define N_EDGES 800000
#define MULC 32
#define FCH 64

// ws float-region offsets (in floats)
#define XS_OFF 0
#define XV_OFF 1600000          // N*32
#define AS_OFF 6400000          // XV + N*96
#define AV_OFF 9600000          // AS + N*64
#define ACC_LEN 17600000        // N*(64+288)
#define H_OFF  24000000         // AV + N*288 : f16 weight region (byte 96e6, 32B aligned)

// f16 weight offsets (in halfs, relative to H_OFF region) -- all 32B aligned
#define W1S_H  0                // K=32,N=32   : 2 frags
#define W1V_H  1024
#define FCW2_H 2048             // K=64,N=160  : 20 frags
#define W2S_H  12288            // K=64,N=64   : 8 frags
#define W2V_H  16384            // K=96,N=32   : 6 frags
#define WSCS_H 19456            // K=512,N=64  : 64 frags
#define WSCV_H 52224            // K=512,N=32  : 32 frags
#define FW1P_H 68608            // fc_w1 zero-padded K=8->32, N=64 : 4 frags
#define H_TOT  70656

__constant__ float INV_NEI = 0.25f;              // 1/sqrt(16)
__constant__ float RS3 = 0.5773502691896258f;    // 1/sqrt(3)
__constant__ float RS2 = 0.7071067811865476f;    // 1/sqrt(2)

__device__ __forceinline__ float silu_f(float x) { return x / (1.0f + __expf(-x)); }

// A-fragment (16x32 f16, M x K) element->K mapping per ISA 7.12.2
__device__ __forceinline__ int a_k(int lane, int e) {
  int j = e >> 1, p = e & 1;
  int kb = (j < 4) ? (2 * j) : (16 + 2 * (j - 4));
  return kb + p + ((lane >> 4) ? 8 : 0);
}

// Swizzled B-fragment source index: [kt][nt][lane][16 halfs] (32B per lane slot)
__device__ __forceinline__ int swz_src(int local, int N) {
  int frag = local >> 9;   // 512 halfs per fragment
  int rem = local & 511;
  int lane = rem >> 4;
  int e = rem & 15;
  int numNT = N >> 4;
  int kt = frag / numNT;
  int nt = frag - kt * numNT;
  int k = kt * 32 + (e & 7) + ((e & 8) << 1) + ((lane >> 4) ? 8 : 0);
  int col = nt * 16 + (lane & 15);
  return k * N + col;
}

// ---------------- weight conversion: f16 + B-fragment swizzle ----------------
__global__ void convert_weights(const float* __restrict__ w1s, const float* __restrict__ w1v,
                                const float* __restrict__ fcw2, const float* __restrict__ w2s,
                                const float* __restrict__ w2v, const float* __restrict__ wscs,
                                const float* __restrict__ wscv, const float* __restrict__ fcw1,
                                _Float16* __restrict__ o) {
  int i = blockIdx.x * blockDim.x + threadIdx.x;
  int stride = gridDim.x * blockDim.x;
  for (int t = i; t < H_TOT; t += stride) {
    int off = t; float v;
    if (off < 1024) v = w1s[swz_src(off, 32)];
    else if ((off -= 1024) < 1024) v = w1v[swz_src(off, 32)];
    else if ((off -= 1024) < 10240) v = fcw2[swz_src(off, 160)];
    else if ((off -= 10240) < 4096) v = w2s[swz_src(off, 64)];
    else if ((off -= 4096) < 3072) v = w2v[swz_src(off, 32)];
    else if ((off -= 3072) < 32768) v = wscs[swz_src(off, 64)];
    else if ((off -= 32768) < 16384) v = wscv[swz_src(off, 32)];
    else {                                  // fc_w1 zero-padded to K=32, 4 fragments
      off -= 16384;
      int rem = off & 511, frag = off >> 9;
      int lane = rem >> 4, e = rem & 15;
      int k = e + ((lane >> 4) ? 16 : 0);
      int colp = frag * 16 + (lane & 15);
      v = (k < 8) ? fcw1[k * FCH + colp] : 0.0f;
    }
    o[t] = (_Float16)v;
  }
}

__global__ void zero_kernel(float* __restrict__ p, int n) {
  int i = blockIdx.x * blockDim.x + threadIdx.x;
  int stride = gridDim.x * blockDim.x;
  for (; i < n; i += stride) p[i] = 0.0f;
}

__device__ __forceinline__ v16h ldfrag(const _Float16* __restrict__ base, int frag, int lane) {
  return *(const v16h*)(base + (((frag << 5) + lane) << 4));
}

// ---------------- node pre: x_s = s@W1_s, x_v = v@W1_v ----------------
__global__ void node_pre(const float* __restrict__ nf, const _Float16* __restrict__ w1s,
                         const _Float16* __restrict__ w1v,
                         float* __restrict__ xs, float* __restrict__ xv) {
  int lane = threadIdx.x & 31;
  int tile = blockIdx.x * (blockDim.x >> 5) + (threadIdx.x >> 5);
  if (tile >= N_NODES / 16) return;
  int base = tile * 16;
  int m = lane & 15, hi = lane >> 4, col = lane & 15;

  // hoisted per-row output base pointers: all stores below use immediate offsets
  float* xsp[8];
  float* xvp[8];
#pragma unroll
  for (int r = 0; r < 8; ++r) {
    int node = base + r + hi * 8;
    xsp[r] = xs + node * MULC + col;
    xvp[r] = xv + node * 96 + col * 3;
  }

  v16h a;
#pragma unroll
  for (int e = 0; e < 16; ++e) {
    int k = a_k(lane, e);
    a[e] = (_Float16)nf[(base + m) * 128 + k];
  }
#pragma unroll
  for (int nt = 0; nt < 2; ++nt) {
    v16h b = ldfrag(w1s, nt, lane);
    v8f c = {};
    c = __builtin_amdgcn_wmma_f32_16x16x32_f16(false, a, false, b, (short)0, c, false, false);
#pragma unroll
    for (int r = 0; r < 8; ++r) xsp[r][nt * 16] = c[r];
  }
#pragma unroll
  for (int cc = 0; cc < 3; ++cc) {
    v16h av;
#pragma unroll
    for (int e = 0; e < 16; ++e) {
      int k = a_k(lane, e);
      av[e] = (_Float16)nf[(base + m) * 128 + 32 + k * 3 + cc];
    }
#pragma unroll
    for (int nt = 0; nt < 2; ++nt) {
      v16h b = ldfrag(w1v, nt, lane);
      v8f c = {};
      c = __builtin_amdgcn_wmma_f32_16x16x32_f16(false, av, false, b, (short)0, c, false, false);
#pragma unroll
      for (int r = 0; r < 8; ++r) xvp[r][nt * 48 + cc] = c[r];
    }
  }
}

// ---------------- fused edge MLP (both layers on WMMA) + messages + scatter ----------------
__global__ void edge_msg(const float* __restrict__ emb, const float* __restrict__ eattr,
                         const int* __restrict__ esrc, const int* __restrict__ edst,
                         const _Float16* __restrict__ fw1p, const float* __restrict__ fcb1,
                         const _Float16* __restrict__ fcw2h,
                         const float* __restrict__ xs, const float* __restrict__ xv,
                         float* __restrict__ a_s, float* __restrict__ a_v) {
  __shared__ _Float16 hbuf[8][16][64];   // per-wave hidden-layer transpose staging (16KB)
  int lane = threadIdx.x & 31;
  int wv = threadIdx.x >> 5;
  int tile = blockIdx.x * (blockDim.x >> 5) + wv;
  if (tile >= N_EDGES / 16) return;
  int base = tile * 16;
  int m = lane & 15, hi = lane >> 4, col = lane & 15;

  // ---- hidden layer as WMMA: (16x8 emb, zero-padded K=32) @ (32x64 fc_w1 padded) ----
  const float4* ep = (const float4*)(emb + (base + m) * 8);
  float4 er0 = ep[0], er1 = ep[1];
  v16h aemb = {};
  aemb[0] = (_Float16)(hi ? 0.0f : er0.x);
  aemb[1] = (_Float16)(hi ? 0.0f : er0.y);
  aemb[2] = (_Float16)(hi ? 0.0f : er0.z);
  aemb[3] = (_Float16)(hi ? 0.0f : er0.w);
  aemb[4] = (_Float16)(hi ? 0.0f : er1.x);
  aemb[5] = (_Float16)(hi ? 0.0f : er1.y);
  aemb[6] = (_Float16)(hi ? 0.0f : er1.z);
  aemb[7] = (_Float16)(hi ? 0.0f : er1.w);

#pragma unroll
  for (int nt = 0; nt < 4; ++nt) {
    v16h b = ldfrag(fw1p, nt, lane);
    v8f c = {};
    c = __builtin_amdgcn_wmma_f32_16x16x32_f16(false, aemb, false, b, (short)0, c, false, false);
    int ch = nt * 16 + col;
    float bias = fcb1[ch];
    // store with bits 3<->4 of ch swapped so A-fragment reads are contiguous
    int pos = (ch & 7) | ((ch & 8) << 1) | ((ch & 16) >> 1) | (ch & 32);
#pragma unroll
    for (int r = 0; r < 8; ++r) {
      hbuf[wv][r + hi * 8][pos] = (_Float16)silu_f(c[r] + bias);
    }
  }
  // same-wave LDS ops are in-order: read back this lane's A-fragments (32B each)
  v16h a0 = *(const v16h*)&hbuf[wv][m][hi * 16];
  v16h a1 = *(const v16h*)&hbuf[wv][m][32 + hi * 16];

  int src8[8], dst8[8];
  float sh0v[8], s1x[8], s1y[8], s1z[8];
#pragma unroll
  for (int r = 0; r < 8; ++r) {
    int ed = base + r + hi * 8;
    src8[r] = esrc[ed];
    dst8[r] = edst[ed];
    float4 ea = ((const float4*)eattr)[ed];
    sh0v[r] = ea.x; s1x[r] = ea.y; s1y[r] = ea.z; s1z[r] = ea.w;
  }

#pragma unroll
  for (int t = 0; t < 10; ++t) {
    v16h b0 = ldfrag(fcw2h, t, lane);        // kt=0
    v16h b1 = ldfrag(fcw2h, 10 + t, lane);   // kt=1
    v8f c = {};
    c = __builtin_amdgcn_wmma_f32_16x16x32_f16(false, a0, false, b0, (short)0, c, false, false);
    c = __builtin_amdgcn_wmma_f32_16x16x32_f16(false, a1, false, b1, (short)0, c, false, false);

    int path = t >> 1;
    int u = (t & 1) * 16 + col;
#pragma unroll
    for (int r = 0; r < 8; ++r) {
      float w = c[r] * INV_NEI;
      int sn = src8[r], dn = dst8[r];
      if (path == 0) {                       // m_s0 = w00 * xs * sh0
        unsafeAtomicAdd(&a_s[dn * 64 + u], w * xs[sn * MULC + u] * sh0v[r]);
      } else if (path == 1) {                // m_v0 = (w01*xs) outer sh1
        float wx = w * xs[sn * MULC + u];
        unsafeAtomicAdd(&a_v[dn * 288 + u * 3 + 0], wx * s1x[r]);
        unsafeAtomicAdd(&a_v[dn * 288 + u * 3 + 1], wx * s1y[r]);
        unsafeAtomicAdd(&a_v[dn * 288 + u * 3 + 2], wx * s1z[r]);
      } else if (path == 2) {                // m_v1 = w10 * xv * sh0
        float ws0 = w * sh0v[r];
        unsafeAtomicAdd(&a_v[dn * 288 + (32 + u) * 3 + 0], ws0 * xv[sn * 96 + u * 3 + 0]);
        unsafeAtomicAdd(&a_v[dn * 288 + (32 + u) * 3 + 1], ws0 * xv[sn * 96 + u * 3 + 1]);
        unsafeAtomicAdd(&a_v[dn * 288 + (32 + u) * 3 + 2], ws0 * xv[sn * 96 + u * 3 + 2]);
      } else if (path == 3) {                // m_s1 = w11s * <xv,sh1>/sqrt3
        float d = xv[sn * 96 + u * 3 + 0] * s1x[r] + xv[sn * 96 + u * 3 + 1] * s1y[r] +
                  xv[sn * 96 + u * 3 + 2] * s1z[r];
        unsafeAtomicAdd(&a_s[dn * 64 + 32 + u], w * d * RS3);
      } else {                               // m_v2 = w11v * cross(xv, sh1)/sqrt2
        float vx = xv[sn * 96 + u * 3 + 0];
        float vy = xv[sn * 96 + u * 3 + 1];
        float vz = xv[sn * 96 + u * 3 + 2];
        float wc = w * RS2;
        unsafeAtomicAdd(&a_v[dn * 288 + (64 + u) * 3 + 0], wc * (vy * s1z[r] - vz * s1y[r]));
        unsafeAtomicAdd(&a_v[dn * 288 + (64 + u) * 3 + 1], wc * (vz * s1x[r] - vx * s1z[r]));
        unsafeAtomicAdd(&a_v[dn * 288 + (64 + u) * 3 + 2], wc * (vx * s1y[r] - vy * s1x[r]));
      }
    }
  }
}

// ---------------- node post: W2 GEMMs + self-connection + gating + residual ----------------
__global__ void node_post(const float* __restrict__ nf, const float* __restrict__ nattr,
                          const float* __restrict__ a_s, const float* __restrict__ a_v,
                          const _Float16* __restrict__ w2sh, const _Float16* __restrict__ w2vh,
                          const _Float16* __restrict__ wscsh, const _Float16* __restrict__ wscvh,
                          float* __restrict__ out) {
  int lane = threadIdx.x & 31;
  int tile = blockIdx.x * (blockDim.x >> 5) + (threadIdx.x >> 5);
  if (tile >= N_NODES / 16) return;
  int base = tile * 16;
  int m = lane & 15, hi = lane >> 4, col = lane & 15;

  v8f ys[4];
#pragma unroll
  for (int t = 0; t < 4; ++t) ys[t] = (v8f){};

  // y_s = a_s @ W2_s   (K=64)
  v16h as0, as1;
#pragma unroll
  for (int e = 0; e < 16; ++e) {
    int k = a_k(lane, e);
    as0[e] = (_Float16)a_s[(base + m) * 64 + k];
    as1[e] = (_Float16)a_s[(base + m) * 64 + 32 + k];
  }
#pragma unroll
  for (int nt = 0; nt < 4; ++nt) {
    v16h b0 = ldfrag(w2sh, nt, lane);
    v16h b1 = ldfrag(w2sh, 4 + nt, lane);
    ys[nt] = __builtin_amdgcn_wmma_f32_16x16x32_f16(false, as0, false, b0, (short)0, ys[nt], false, false);
    ys[nt] = __builtin_amdgcn_wmma_f32_16x16x32_f16(false, as1, false, b1, (short)0, ys[nt], false, false);
  }

  // lane-fixed attr permutation, converted to f16 once; per-K-step builds are pure f16 muls
  float atr[16];
#pragma unroll
  for (int aa = 0; aa < 16; ++aa) atr[aa] = nattr[(base + m) * 16 + aa];
  v16h hatr;
#pragma unroll
  for (int e = 0; e < 16; ++e) hatr[e] = (_Float16)atr[a_k(lane, e) & 15];

  // self-connection scalar: (s ⊗ attrs) (16x512) @ Wsc_s (512x64)
#pragma unroll
  for (int ks = 0; ks < 16; ++ks) {
    _Float16 hs0 = (_Float16)nf[(base + m) * 128 + 2 * ks];
    _Float16 hs1 = (_Float16)nf[(base + m) * 128 + 2 * ks + 1];
    v16h af;
#pragma unroll
    for (int e = 0; e < 16; ++e) af[e] = hatr[e] * (e < 8 ? hs0 : hs1);
#pragma unroll
    for (int nt = 0; nt < 4; ++nt) {
      v16h b = ldfrag(wscsh, ks * 4 + nt, lane);
      ys[nt] = __builtin_amdgcn_wmma_f32_16x16x32_f16(false, af, false, b, (short)0, ys[nt], false, false);
    }
  }

  // y_v per component: a_v @ W2_v (K=96) + (v ⊗ attrs) @ Wsc_v (K=512)
  v8f yv[3][2];
#pragma unroll
  for (int cc = 0; cc < 3; ++cc)
#pragma unroll
    for (int nt = 0; nt < 2; ++nt) yv[cc][nt] = (v8f){};

#pragma unroll
  for (int cc = 0; cc < 3; ++cc) {
#pragma unroll
    for (int ks = 0; ks < 3; ++ks) {
      v16h af;
#pragma unroll
      for (int e = 0; e < 16; ++e) {
        int k = ks * 32 + a_k(lane, e);
        af[e] = (_Float16)a_v[(base + m) * 288 + k * 3 + cc];
      }
#pragma unroll
      for (int nt = 0; nt < 2; ++nt) {
        v16h b = ldfrag(w2vh, ks * 2 + nt, lane);
        yv[cc][nt] = __builtin_amdgcn_wmma_f32_16x16x32_f16(false, af, false, b, (short)0, yv[cc][nt], false, false);
      }
    }
#pragma unroll
    for (int ks = 0; ks < 16; ++ks) {
      _Float16 hv0 = (_Float16)nf[(base + m) * 128 + 32 + (2 * ks) * 3 + cc];
      _Float16 hv1 = (_Float16)nf[(base + m) * 128 + 32 + (2 * ks + 1) * 3 + cc];
      v16h af;
#pragma unroll
      for (int e = 0; e < 16; ++e) af[e] = hatr[e] * (e < 8 ? hv0 : hv1);
#pragma unroll
      for (int nt = 0; nt < 2; ++nt) {
        v16h b = ldfrag(wscvh, ks * 2 + nt, lane);
        yv[cc][nt] = __builtin_amdgcn_wmma_f32_16x16x32_f16(false, af, false, b, (short)0, yv[cc][nt], false, false);
      }
    }
  }

  // gating + residual, direct store to out
#pragma unroll
  for (int r = 0; r < 8; ++r) {
    int node = base + r + hi * 8;
    const float* nfr = nf + node * 128;
    float* op = out + node * 128;
    op[col] = nfr[col] + silu_f(ys[0][r]);
    op[16 + col] = nfr[16 + col] + silu_f(ys[1][r]);
    float g0 = silu_f(ys[2][r]);
    float g1 = silu_f(ys[3][r]);
#pragma unroll
    for (int cc = 0; cc < 3; ++cc) {
      op[32 + col * 3 + cc] = nfr[32 + col * 3 + cc] + yv[cc][0][r] * g0;
      op[32 + (16 + col) * 3 + cc] = nfr[32 + (16 + col) * 3 + cc] + yv[cc][1][r] * g1;
    }
  }
}

extern "C" void kernel_launch(void* const* d_in, const int* in_sizes, int n_in,
                              void* d_out, int out_size, void* d_ws, size_t ws_size,
                              hipStream_t stream) {
  const float* node_feats = (const float*)d_in[0];
  const float* node_attrs = (const float*)d_in[1];
  const float* edge_emb = (const float*)d_in[2];
  const float* edge_attrs = (const float*)d_in[3];
  const int* edge_src = (const int*)d_in[4];
  const int* edge_dst = (const int*)d_in[5];
  const float* W1_s = (const float*)d_in[6];
  const float* W1_v = (const float*)d_in[7];
  const float* fc_w1 = (const float*)d_in[8];
  const float* fc_b1 = (const float*)d_in[9];
  const float* fc_w2 = (const float*)d_in[10];
  const float* W2_s = (const float*)d_in[11];
  const float* W2_v = (const float*)d_in[12];
  const float* Wsc_s = (const float*)d_in[13];
  const float* Wsc_v = (const float*)d_in[14];
  float* out = (float*)d_out;
  float* ws = (float*)d_ws;

  float* xs = ws + XS_OFF;
  float* xv = ws + XV_OFF;
  float* a_s = ws + AS_OFF;
  float* a_v = ws + AV_OFF;
  _Float16* hw = (_Float16*)(ws + H_OFF);

  convert_weights<<<128, 256, 0, stream>>>(W1_s, W1_v, fc_w2, W2_s, W2_v, Wsc_s, Wsc_v,
                                           fc_w1, hw);
  zero_kernel<<<4096, 256, 0, stream>>>(a_s, ACC_LEN);

  int node_blocks = (N_NODES / 16 + 7) / 8;   // 8 wave-tiles per 256-thread block
  node_pre<<<node_blocks, 256, 0, stream>>>(node_feats, hw + W1S_H, hw + W1V_H, xs, xv);

  int edge_blocks = (N_EDGES / 16) / 8;       // 50000 wave-tiles
  edge_msg<<<edge_blocks, 256, 0, stream>>>(edge_emb, edge_attrs, edge_src, edge_dst,
                                            hw + FW1P_H, fc_b1, hw + FCW2_H, xs, xv, a_s, a_v);

  node_post<<<node_blocks, 256, 0, stream>>>(node_feats, node_attrs, a_s, a_v,
                                             hw + W2S_H, hw + W2V_H, hw + WSCS_H, hw + WSCV_H,
                                             out);
}
